// Sequence_14877766713698
// MI455X (gfx1250) — compile-verified, matrix operand order checked by reading, not run on hardware
//
#include <hip/hip_runtime.h>
#include <hip/hip_bf16.h>
#include <stdint.h>

// Problem constants (from reference)
#define T_STEPS 512
#define BATCH   4096
#define IN_DIM  11
#define BT      32          // batch rows per workgroup (2 WMMA M-tiles)
#define NTHREADS 256        // 8 waves (wave32)

// bf16 weight region layout inside d_ws / LDS (ushort units)
#define OFF_WIH1 0          // K=32(pad from 11), N=512 : 1x32 blocks
#define OFF_WHH1 16384      // K=128, N=512 : 4x32 blocks
#define OFF_WIH2 81920      // K=128, N=256 : 4x16 blocks
#define OFF_WHH2 114688     // K=64,  N=256 : 2x16 blocks
#define OFF_WIH3 131072     // K=64,  N=128 : 2x8  blocks
#define OFF_WHH3 139264     // K=32,  N=128 : 1x8  blocks
#define W_TOTAL_US 143360   // 280 KB of bf16 weights

typedef __attribute__((ext_vector_type(16))) __bf16 v16bf;
typedef __attribute__((ext_vector_type(8)))  float  v8f;

union ABu { uint4 q[2]; v16bf v; };

__device__ __forceinline__ unsigned short f2bf(float f) {
    unsigned u = __float_as_uint(f);
    u += 0x7FFFu + ((u >> 16) & 1u);          // round-to-nearest-even
    return (unsigned short)(u >> 16);
}
__device__ __forceinline__ float bf2f(unsigned short s) {
    return __uint_as_float(((unsigned)s) << 16);
}
__device__ __forceinline__ float fsigmoid(float x) {
    return 1.0f / (1.0f + __expf(-x));
}
__device__ __forceinline__ float ftanh(float x) {
    float t = __expf(-2.0f * fabsf(x));
    float r = (1.0f - t) / (1.0f + t);
    return copysignf(r, x);
}

// ---------------------------------------------------------------------------
// Kernel 1: convert f32 weights -> bf16, pre-swizzled into WMMA B-operand
// layout. One 32x16 (KxN) block = 512 bf16, stored lane-major: lane L holds 16
// contiguous bf16 (its 8 B-VGPRs). Element mapping per CDNA5 ISA 16-bit layout:
//   k = (v>=4 ? 16 : 0) + (L>>4)*8 + (v&3)*2 + p,  n = ntile*16 + (L&15)
// ---------------------------------------------------------------------------
__global__ void convert_weights(const float* __restrict__ Wih1, const float* __restrict__ Whh1,
                                const float* __restrict__ Wih2, const float* __restrict__ Whh2,
                                const float* __restrict__ Wih3, const float* __restrict__ Whh3,
                                unsigned short* __restrict__ ws)
{
    int b = blockIdx.x;
    const float* W; int Ksrc, NT, dstoff, local;
    if      (b < 32)  { W = Wih1; Ksrc = 11;  NT = 32; dstoff = OFF_WIH1; local = b;       }
    else if (b < 160) { W = Whh1; Ksrc = 128; NT = 32; dstoff = OFF_WHH1; local = b - 32;  }
    else if (b < 224) { W = Wih2; Ksrc = 128; NT = 16; dstoff = OFF_WIH2; local = b - 160; }
    else if (b < 256) { W = Whh2; Ksrc = 64;  NT = 16; dstoff = OFF_WHH2; local = b - 224; }
    else if (b < 272) { W = Wih3; Ksrc = 64;  NT = 8;  dstoff = OFF_WIH3; local = b - 256; }
    else              { W = Whh3; Ksrc = 32;  NT = 8;  dstoff = OFF_WHH3; local = b - 272; }
    int kt = local / NT, nt = local % NT;
    for (int e = threadIdx.x; e < 512; e += blockDim.x) {
        int lane = e >> 4, idx = e & 15;
        int v = idx >> 1, p = idx & 1;
        int kl = ((v & 4) ? 16 : 0) + ((lane >> 4) << 3) + ((v & 3) << 1) + p;
        int k  = kt * 32 + kl;
        int n  = nt * 16 + (lane & 15);
        float val = (k < Ksrc) ? W[n * Ksrc + k] : 0.0f;  // zero-pad K (IN=11 -> 32)
        ws[dstoff + local * 512 + e] = f2bf(val);
    }
}

// ---------------------------------------------------------------------------
// Per-wave LSTM gate unit: MT M-tiles x one hidden 16-slice; 4 gate WMMA
// accumulators each, K-loops over the input activation (A1) and own h (A2).
// c-state lives in registers in accumulator layout. h result returned as bf16
// (stored to LDS by caller after a barrier, to avoid RAW on the h buffers).
// ---------------------------------------------------------------------------
template<int MT, int K1T, int K2T, int HT>
__device__ __forceinline__ void lstm_units(
    const unsigned short* sA1, int strideA1,
    const unsigned short* sA2, int strideA2,
    const unsigned short* sWihL, const unsigned short* sWhhL,
    const float* sB, int H,
    float cr[][8], unsigned short hr[][8],
    int mt0, int ht, int lane)
{
    v8f acc[MT][4] = {};
    const int col   = lane & 15;
    const int khalf = (lane >> 4) << 3;
    const int NT = 4 * HT;

#pragma unroll
    for (int kt = 0; kt < K1T; ++kt) {
        ABu a[MT];
#pragma unroll
        for (int m = 0; m < MT; ++m) {
            const unsigned short* pa = sA1 + ((mt0 + m) * 16 + col) * strideA1 + kt * 32 + khalf;
            a[m].q[0] = *(const uint4*)(pa);       // K = khalf .. khalf+7
            a[m].q[1] = *(const uint4*)(pa + 16);  // K = 16+khalf .. 16+khalf+7
        }
#pragma unroll
        for (int qg = 0; qg < 4; ++qg) {
            ABu bm;
            const unsigned short* pb = sWihL + (kt * NT + qg * HT + ht) * 512 + lane * 16;
            bm.q[0] = *(const uint4*)(pb);
            bm.q[1] = *(const uint4*)(pb + 8);
#pragma unroll
            for (int m = 0; m < MT; ++m)
                acc[m][qg] = __builtin_amdgcn_wmma_f32_16x16x32_bf16(
                    false, a[m].v, false, bm.v, (short)0, acc[m][qg], false, false);
        }
    }
#pragma unroll
    for (int kt = 0; kt < K2T; ++kt) {
        ABu a[MT];
#pragma unroll
        for (int m = 0; m < MT; ++m) {
            const unsigned short* pa = sA2 + ((mt0 + m) * 16 + col) * strideA2 + kt * 32 + khalf;
            a[m].q[0] = *(const uint4*)(pa);
            a[m].q[1] = *(const uint4*)(pa + 16);
        }
#pragma unroll
        for (int qg = 0; qg < 4; ++qg) {
            ABu bm;
            const unsigned short* pb = sWhhL + (kt * NT + qg * HT + ht) * 512 + lane * 16;
            bm.q[0] = *(const uint4*)(pb);
            bm.q[1] = *(const uint4*)(pb + 8);
#pragma unroll
            for (int m = 0; m < MT; ++m)
                acc[m][qg] = __builtin_amdgcn_wmma_f32_16x16x32_bf16(
                    false, a[m].v, false, bm.v, (short)0, acc[m][qg], false, false);
        }
    }

    const int hidx = ht * 16 + col;
    const float bi  = sB[hidx];
    const float bff = sB[H + hidx];
    const float bg  = sB[2 * H + hidx];
    const float bo  = sB[3 * H + hidx];
#pragma unroll
    for (int m = 0; m < MT; ++m) {
#pragma unroll
        for (int r = 0; r < 8; ++r) {
            float iv = fsigmoid(acc[m][0][r] + bi);
            float fv = fsigmoid(acc[m][1][r] + bff);
            float gv = ftanh(acc[m][2][r] + bg);
            float ov = fsigmoid(acc[m][3][r] + bo);
            float c  = fv * cr[m][r] + iv * gv;
            cr[m][r] = c;
            hr[m][r] = f2bf(ov * ftanh(c));
        }
    }
}

template<int MT>
__device__ __forceinline__ void store_h(unsigned short* sH, int strideH,
                                        unsigned short hr[][8], int mt0, int ht, int lane)
{
    const int hidx = ht * 16 + (lane & 15);
#pragma unroll
    for (int m = 0; m < MT; ++m) {
        const int mbase = (mt0 + m) * 16 + ((lane >> 4) << 3);  // C/D layout: lanes 16-31 = M+8
#pragma unroll
        for (int r = 0; r < 8; ++r)
            sH[(mbase + r) * strideH + hidx] = hr[m][r];
    }
}

// ---------------------------------------------------------------------------
// Kernel 2: persistent 3-layer LSTM. 128 WGs x 32 batch rows; all weights
// LDS-resident (≈300 KB of CDNA5's 320 KB WGP LDS), 513 sequential steps.
// ---------------------------------------------------------------------------
__global__ void __launch_bounds__(NTHREADS)
lstm_persistent(const float* __restrict__ input,
                const unsigned short* __restrict__ ws,
                const float* __restrict__ b_ih1, const float* __restrict__ b_hh1,
                const float* __restrict__ b_ih2, const float* __restrict__ b_hh2,
                const float* __restrict__ b_ih3, const float* __restrict__ b_hh3,
                const float* __restrict__ fc1_w, const float* __restrict__ fc1_b,
                const float* __restrict__ fc2_w, const float* __restrict__ fc2_b,
                float* __restrict__ out)
{
    extern __shared__ __align__(16) char smem[];
    unsigned short* sW   = (unsigned short*)smem;          // 143360 us (280 KB)
    unsigned short* sX   = sW + W_TOTAL_US;                // 32x32 bf16 (zero-padded K)
    unsigned short* sH1v = sX + 1024;                      // 32x128 bf16
    unsigned short* sH2v = sH1v + 4096;                    // 32x64  bf16
    unsigned short* sH3v = sH2v + 2048;                    // 32x32  bf16
    float* sB1 = (float*)(sH3v + 1024);                    // 512 f32 (bih+bhh)
    float* sB2 = sB1 + 512;                                // 256 f32
    float* sB3 = sB2 + 256;                                // 128 f32

    const int tid  = threadIdx.x;
    const int wave = tid >> 5;
    const int lane = tid & 31;
    const int batch0 = blockIdx.x * BT;

    // Stage all bf16 weights global -> LDS as 128-bit transfers (once).
    {
        const uint4* src = (const uint4*)ws;
        uint4* dst = (uint4*)sW;
        for (int i = tid; i < W_TOTAL_US / 8; i += NTHREADS) dst[i] = src[i];
    }
    for (int i = tid; i < 512;  i += NTHREADS) sB1[i] = b_ih1[i] + b_hh1[i];
    for (int i = tid; i < 256;  i += NTHREADS) sB2[i] = b_ih2[i] + b_hh2[i];
    for (int i = tid; i < 128;  i += NTHREADS) sB3[i] = b_ih3[i] + b_hh3[i];
    for (int i = tid; i < 1024; i += NTHREADS) sX[i]   = 0;   // K-pad stays zero forever
    for (int i = tid; i < 4096; i += NTHREADS) sH1v[i] = 0;
    for (int i = tid; i < 2048; i += NTHREADS) sH2v[i] = 0;
    for (int i = tid; i < 1024; i += NTHREADS) sH3v[i] = 0;

    float c1r[2][8], c2r[1][8], c3r[1][8];
#pragma unroll
    for (int r = 0; r < 8; ++r) { c1r[0][r] = 0.f; c1r[1][r] = 0.f; c2r[0][r] = 0.f; c3r[0][r] = 0.f; }

    __syncthreads();

    for (int step = 0; step <= T_STEPS; ++step) {
        const int t = (step < T_STEPS) ? step : 4;   // extra step on input[4]

        // Stage x_t (32 x 11 f32) -> bf16 LDS tile.
        const float* xsrc = input + ((size_t)t * BATCH + batch0) * IN_DIM;
        for (int e = tid; e < BT * IN_DIM; e += NTHREADS) {
            int row = e / IN_DIM, colx = e % IN_DIM;
            sX[row * 32 + colx] = f2bf(xsrc[e]);
        }
        // Warm next step's x tile in cache (global_prefetch_b8).
        if (step < T_STEPS) {
            int tn = (step + 1 < T_STEPS) ? (step + 1) : 4;
            const char* pf = (const char*)(input + ((size_t)tn * BATCH + batch0) * IN_DIM);
            if (tid < 11) __builtin_prefetch(pf + tid * 128, 0, 0);
        }
        __syncthreads();

        // Layer 1: H=128 -> 8 hidden tiles, each wave does both M-tiles of its slice.
        unsigned short h1r[2][8];
        lstm_units<2, 1, 4, 8>(sX, 32, sH1v, 128,
                               sW + OFF_WIH1, sW + OFF_WHH1, sB1, 128,
                               c1r, h1r, 0, wave, lane);
        __syncthreads();
        store_h<2>(sH1v, 128, h1r, 0, wave, lane);
        __syncthreads();

        // Layer 2: H=64 -> 4 hidden tiles x 2 M-tiles = 8 units, one per wave.
        unsigned short h2r[1][8];
        lstm_units<1, 4, 2, 4>(sH1v, 128, sH2v, 64,
                               sW + OFF_WIH2, sW + OFF_WHH2, sB2, 64,
                               c2r, h2r, wave >> 2, wave & 3, lane);
        __syncthreads();
        store_h<1>(sH2v, 64, h2r, wave >> 2, wave & 3, lane);
        __syncthreads();

        // Layer 3: H=32 -> 2 hidden tiles x 2 M-tiles = 4 units, waves 0-3.
        unsigned short h3r[1][8];
        if (wave < 4)
            lstm_units<1, 2, 1, 2>(sH2v, 64, sH3v, 32,
                                   sW + OFF_WIH3, sW + OFF_WHH3, sB3, 32,
                                   c3r, h3r, wave >> 1, wave & 1, lane);
        __syncthreads();
        if (wave < 4) store_h<1>(sH3v, 32, h3r, wave >> 1, wave & 1, lane);
        __syncthreads();
    }

    // Linear head: out = (h3 @ fc1_w.T + fc1_b) @ fc2_w.T + fc2_b (no nonlinearity).
    if (tid < BT) {
        float acc2 = fc2_b[0];
#pragma unroll
        for (int j = 0; j < 16; ++j) {
            float s = fc1_b[j];
#pragma unroll
            for (int k = 0; k < 32; ++k)
                s += bf2f(sH3v[tid * 32 + k]) * fc1_w[j * 32 + k];
            acc2 += s * fc2_w[j];
        }
        out[batch0 + tid] = acc2;
    }
}

extern "C" void kernel_launch(void* const* d_in, const int* in_sizes, int n_in,
                              void* d_out, int out_size, void* d_ws, size_t ws_size,
                              hipStream_t stream)
{
    const float* input = (const float*)d_in[0];
    const float* Wih1  = (const float*)d_in[1];
    const float* Whh1  = (const float*)d_in[2];
    const float* bih1  = (const float*)d_in[3];
    const float* bhh1  = (const float*)d_in[4];
    const float* Wih2  = (const float*)d_in[5];
    const float* Whh2  = (const float*)d_in[6];
    const float* bih2  = (const float*)d_in[7];
    const float* bhh2  = (const float*)d_in[8];
    const float* Wih3  = (const float*)d_in[9];
    const float* Whh3  = (const float*)d_in[10];
    const float* bih3  = (const float*)d_in[11];
    const float* bhh3  = (const float*)d_in[12];
    const float* fc1w  = (const float*)d_in[13];
    const float* fc1b  = (const float*)d_in[14];
    const float* fc2w  = (const float*)d_in[15];
    const float* fc2b  = (const float*)d_in[16];
    unsigned short* wsW = (unsigned short*)d_ws;
    float* out = (float*)d_out;

    // 1) bf16-convert + swizzle weights into workspace (280 blocks of 32x16).
    convert_weights<<<280, 256, 0, stream>>>(Wih1, Whh1, Wih2, Whh2, Wih3, Whh3, wsW);

    // 2) Persistent LSTM: 128 WGs x 256 threads, ~300 KB dynamic LDS per WGP.
    const size_t shmem = (size_t)(W_TOTAL_US + 1024 + 4096 + 2048 + 1024) * 2
                       + (size_t)(512 + 256 + 128) * 4;   // = 306688 B
    lstm_persistent<<<BATCH / BT, NTHREADS, shmem, stream>>>(
        input, wsW, bih1, bhh1, bih2, bhh2, bih3, bhh3,
        fc1w, fc1b, fc2w, fc2b, out);
}
